// CrossChannelAttention_64355789963669
// MI455X (gfx1250) — matrix operand order repaired and checked by reading, not compiled, not run on hardware
//
#include <hip/hip_runtime.h>
#include <hip/hip_bf16.h>

// ---------------------------------------------------------------------------
// CrossChannelAttention transformer block for MI455X (gfx1250, wave32).
// Big GEMMs (QKV, O, FFN) run on v_wmma_f32_16x16x32_f16 with f32 accum.
// Weights are pre-transposed to f16 so both A and B tiles stage as contiguous
// 16B copies; staging is double-buffered with global_load_async_to_lds_b128
// (ASYNCcnt) when the toolchain exposes it.
// ---------------------------------------------------------------------------

typedef __attribute__((ext_vector_type(16))) _Float16     v16h;
typedef __attribute__((ext_vector_type(8)))  float        v8f;
typedef __attribute__((ext_vector_type(8)))  unsigned int v8u;
typedef int cca_v4i __attribute__((vector_size(16)));     // matches builtin's pointee type

#define BDIM 4096
#define CDIM 16
#define HID  1024
#define NH   16
#define HD   64
#define NCT  32
#define MROWS (BDIM * CDIM)   // 65536

#if __has_builtin(__builtin_amdgcn_global_load_async_to_lds_b128)
#define CCA_ASYNC_LDS 1
#endif

// 16-byte global -> LDS copy (async on CDNA5 when available)
__device__ __forceinline__ void cca_cp16(const void* g, void* l) {
#ifdef CCA_ASYNC_LDS
    __builtin_amdgcn_global_load_async_to_lds_b128(
        (__attribute__((address_space(1))) cca_v4i*)(unsigned long long)g,
        (__attribute__((address_space(3))) cca_v4i*)(unsigned int)(unsigned long long)l,
        0, 0);
#else
    *(uint4*)l = *(const uint4*)g;
#endif
}

__device__ __forceinline__ void cca_async_wait0() {
#ifdef CCA_ASYNC_LDS
#if __has_builtin(__builtin_amdgcn_s_wait_asynccnt)
    __builtin_amdgcn_s_wait_asynccnt(0);
#else
    asm volatile("s_wait_asynccnt 0x0" ::: "memory");
#endif
#endif
}

// ---------------------------------------------------------------------------
// fp32 (K x N) -> fp16 transposed (N x K), LDS-tiled so both sides coalesce
// ---------------------------------------------------------------------------
__global__ __launch_bounds__(256)
void cca_convert_transpose_f16(const float* __restrict__ in, _Float16* __restrict__ out,
                               int K, int N) {
    __shared__ float t[32][33];
    const int kb = blockIdx.x * 32;
    const int nb = blockIdx.y * 32;
    const int tx = threadIdx.x & 31;
    const int ty = threadIdx.x >> 5;   // 32 x 8 threads
#pragma unroll
    for (int i = 0; i < 32; i += 8)
        t[ty + i][tx] = in[(size_t)(kb + ty + i) * N + nb + tx];
    __syncthreads();
#pragma unroll
    for (int i = 0; i < 32; i += 8)
        out[(size_t)(nb + ty + i) * K + kb + tx] = (_Float16)t[tx][ty + i];
}

// ---------------------------------------------------------------------------
// LayerNorm over HID=1024, one block (256 threads) per row, output fp16
// ---------------------------------------------------------------------------
__global__ __launch_bounds__(256)
void cca_layernorm_f16(const float* __restrict__ x,
                       const float* __restrict__ gamma,
                       const float* __restrict__ beta,
                       _Float16* __restrict__ out) {
    const int row = blockIdx.x;
    const int tid = threadIdx.x;
    const float* xr = x + (size_t)row * HID;

    __shared__ float redA[256];
    __shared__ float redB[256];

    float vloc[4];
    float s = 0.f, s2 = 0.f;
#pragma unroll
    for (int i = 0; i < 4; ++i) {
        float t = xr[tid + i * 256];
        vloc[i] = t;
        s  += t;
        s2 += t * t;
    }
    redA[tid] = s;
    redB[tid] = s2;
    __syncthreads();
    for (int off = 128; off > 0; off >>= 1) {
        if (tid < off) { redA[tid] += redA[tid + off]; redB[tid] += redB[tid + off]; }
        __syncthreads();
    }
    const float mean = redA[0] * (1.f / HID);
    const float var  = redB[0] * (1.f / HID) - mean * mean;
    const float inv  = rsqrtf(var + 1e-5f);

    _Float16* orow = out + (size_t)row * HID;
#pragma unroll
    for (int i = 0; i < 4; ++i) {
        int c = tid + i * 256;
        orow[c] = (_Float16)((vloc[i] - mean) * inv * gamma[c] + beta[c]);
    }
}

// ---------------------------------------------------------------------------
// WMMA GEMM: Out(M x N) = A(M x K, f16 row-major) * Bt^T, with Bt given as
// (N x K, f16 row-major) pre-transposed weights. Block tile 128x128, BK=32,
// 256 threads = 8 waves, each wave a 32x64 sub-tile (2x4 16x16 accumulators).
// Double-buffered LDS; async global->LDS staging.
// Epilogues:
//   0: out_f16 = acc + bias
//   1: out_f16 = acc + bias + ct_key_emb[ctype[m/16]]   (K projection)
//   2: out_f32 = acc + bias + resid                     (Wo / W2 residual)
//   3: out_f16 = elu(acc + bias)                        (FFN first layer)
// ---------------------------------------------------------------------------
template<int EPI>
__global__ __launch_bounds__(256)
void cca_gemm_wmma(const _Float16* __restrict__ A,
                   const _Float16* __restrict__ Bt,
                   const float* __restrict__ bias,
                   void* __restrict__ Out,
                   const float* __restrict__ resid,
                   const float* __restrict__ ctkey,
                   const int* __restrict__ ctype,
                   int Nsz, int Ksz) {
    constexpr int BM = 128, BN = 128, BK = 32;
    constexpr int LDT = 40;                          // LDS row pitch in ushorts (80B)
    __shared__ unsigned short As[2][BM * LDT];       // A tile,  row m : 32 K-values
    __shared__ unsigned short Bs[2][BN * LDT];       // B tile,  row n : 32 K-values

    const int tid  = threadIdx.x;
    const int lane = tid & 31;
    const int wave = tid >> 5;
    const int wm = wave & 3;                         // 4 waves along M (32 rows each)
    const int wn = wave >> 2;                        // 2 waves along N (64 cols each)
    const int lm = lane & 15;
    const int lh = lane >> 4;
    const int m0 = blockIdx.y * BM;
    const int n0 = blockIdx.x * BN;

    // staging: 512 16B chunks per tile, 256 threads -> 2 chunks per tile each
    const int srow = tid >> 2;                       // 0..63 base row, +64 on 2nd chunk
    const int sq   = tid & 3;                        // 16B chunk within a 64B row
    auto stage = [&](int kt, int buf) {
        const size_t kof = (size_t)kt * BK + sq * 8;
#pragma unroll
        for (int it = 0; it < 2; ++it) {
            int row = srow + it * 64;
            cca_cp16(A  + (size_t)(m0 + row) * Ksz + kof, &As[buf][row * LDT + sq * 8]);
            cca_cp16(Bt + (size_t)(n0 + row) * Ksz + kof, &Bs[buf][row * LDT + sq * 8]);
        }
    };

    v8f acc[2][4];
#pragma unroll
    for (int i = 0; i < 2; ++i)
#pragma unroll
        for (int j = 0; j < 4; ++j)
#pragma unroll
            for (int e = 0; e < 8; ++e) acc[i][j][e] = 0.f;

    const int nk = Ksz / BK;
    stage(0, 0);
    for (int kt = 0; kt < nk; ++kt) {
        const int cur = kt & 1;
        cca_async_wait0();
        __syncthreads();                             // tile kt resident in buf 'cur'
        if (kt + 1 < nk) stage(kt + 1, cur ^ 1);     // prefetch next tile (async)

        // ---- fragments (contiguous k = 16h+s, same order for A and B): 2x ds_load_b128 each ----
        v16h afr[2], bfr[4];
#pragma unroll
        for (int mt = 0; mt < 2; ++mt) {
            const unsigned short* ap = &As[cur][(wm * 32 + mt * 16 + lm) * LDT];
            uint4 lo = *(const uint4*)(ap + 16 * lh);      // K = 16h..16h+7   (VGPR 0-3)
            uint4 hi = *(const uint4*)(ap + 16 * lh + 8);  // K = 16h+8..16h+15 (VGPR 4-7)
            v8u u = { lo.x, lo.y, lo.z, lo.w, hi.x, hi.y, hi.z, hi.w };
            afr[mt] = __builtin_bit_cast(v16h, u);
        }
#pragma unroll
        for (int nt = 0; nt < 4; ++nt) {
            const unsigned short* bp = &Bs[cur][(wn * 64 + nt * 16 + lm) * LDT + 16 * lh];
            uint4 lo = *(const uint4*)(bp);                // K = 16h..16h+7
            uint4 hi = *(const uint4*)(bp + 8);            // K = 16h+8..16h+15
            v8u u = { lo.x, lo.y, lo.z, lo.w, hi.x, hi.y, hi.z, hi.w };
            bfr[nt] = __builtin_bit_cast(v16h, u);
        }

#pragma unroll
        for (int mt = 0; mt < 2; ++mt)
#pragma unroll
            for (int nt = 0; nt < 4; ++nt)
                acc[mt][nt] = __builtin_amdgcn_wmma_f32_16x16x32_f16(
                    false, afr[mt], false, bfr[nt], (short)0, acc[mt][nt], false, false);

        __syncthreads();                             // all reads of 'cur' done
    }

    // ---- epilogue: lane (lm,lh) owns N=lm, M=r+8*lh of each 16x16 tile ----
#pragma unroll
    for (int mt = 0; mt < 2; ++mt) {
        const int tileM = m0 + wm * 32 + mt * 16;    // one 16-row tile == one batch (C=16)
        const float* ck = nullptr;
        if constexpr (EPI == 1) ck = ctkey + (size_t)ctype[tileM >> 4] * HID;
        int   gnv[4];
        float bnv[4];
#pragma unroll
        for (int nt = 0; nt < 4; ++nt) {
            int gn = n0 + wn * 64 + nt * 16 + lm;
            gnv[nt] = gn;
            float bn = bias[gn];
            if constexpr (EPI == 1) bn += ck[gn];
            bnv[nt] = bn;
        }
#pragma unroll
        for (int r = 0; r < 8; ++r) {
            const size_t rb = (size_t)(tileM + 8 * lh + r) * Nsz;
#pragma unroll
            for (int nt = 0; nt < 4; ++nt) {
                float v = acc[mt][nt][r] + bnv[nt];
                size_t idx = rb + gnv[nt];
                if constexpr (EPI == 0 || EPI == 1) {
                    ((_Float16*)Out)[idx] = (_Float16)v;
                } else if constexpr (EPI == 2) {
                    ((float*)Out)[idx] = v + resid[idx];
                } else { // EPI == 3: ELU
                    float e = v > 0.f ? v : (__expf(v) - 1.f);
                    ((_Float16*)Out)[idx] = (_Float16)e;
                }
            }
        }
    }
}

// ---------------------------------------------------------------------------
// Attention core: one block per (batch, head). C=16, HD=64 -> tiny tiles.
// scores = qk^T/8 + 0.1*ct_bias ; mask inactive keys ; softmax ; out = p@v
// ---------------------------------------------------------------------------
__global__ __launch_bounds__(256)
void cca_attention(const _Float16* __restrict__ q,
                   const _Float16* __restrict__ k,
                   const _Float16* __restrict__ v,
                   const float* __restrict__ ct_bias_emb,
                   const int* __restrict__ ctype,
                   const int* __restrict__ active,
                   _Float16* __restrict__ out) {
    const int bh = blockIdx.x;
    const int b = bh >> 4;
    const int h = bh & 15;
    const int tid = threadIdx.x;
    const int ct = ctype[b];

    __shared__ float qs[CDIM * HD];
    __shared__ float ks[CDIM * HD];
    __shared__ float vs[CDIM * HD];
    __shared__ float sc[CDIM * CDIM];

    const size_t base = ((size_t)b * CDIM) * HID + (size_t)h * HD;
    for (int i = tid; i < CDIM * HD; i += 256) {
        int ci = i >> 6, d = i & 63;
        size_t g = base + (size_t)ci * HID + d;
        qs[i] = (float)q[g];
        ks[i] = (float)k[g];          // ct_key_emb already fused into K projection
        vs[i] = (float)v[g];
    }
    __syncthreads();

    {   // 256 threads <-> 256 (qi,ki) score entries
        int qi = tid >> 4, ki = tid & 15;
        float s = 0.f;
#pragma unroll
        for (int d = 0; d < HD; ++d) s += qs[qi * HD + d] * ks[ki * HD + d];
        s = s * 0.125f + 0.1f * ct_bias_emb[(size_t)ct * (CDIM * CDIM) + qi * CDIM + ki];
        if (active[b * CDIM + ki] == 0) s = -1e30f;
        sc[tid] = s;
    }
    __syncthreads();

    if (tid < CDIM) {   // per-row softmax (row length 16)
        float mx = -1e38f;
#pragma unroll
        for (int ki = 0; ki < CDIM; ++ki) mx = fmaxf(mx, sc[tid * CDIM + ki]);
        if (mx <= -1e29f) {            // fully-masked row -> zeros (nan_to_num)
#pragma unroll
            for (int ki = 0; ki < CDIM; ++ki) sc[tid * CDIM + ki] = 0.f;
        } else {
            float sum = 0.f;
#pragma unroll
            for (int ki = 0; ki < CDIM; ++ki) {
                float e = __expf(sc[tid * CDIM + ki] - mx);
                sc[tid * CDIM + ki] = e;
                sum += e;
            }
            float inv = 1.f / sum;
#pragma unroll
            for (int ki = 0; ki < CDIM; ++ki) sc[tid * CDIM + ki] *= inv;
        }
    }
    __syncthreads();

    for (int i = tid; i < CDIM * HD; i += 256) {
        int qi = i >> 6, d = i & 63;
        float o = 0.f;
#pragma unroll
        for (int ki = 0; ki < CDIM; ++ki) o += sc[qi * CDIM + ki] * vs[ki * HD + d];
        out[base + (size_t)qi * HID + d] = (_Float16)o;
    }
}

// ---------------------------------------------------------------------------
// launch
// ---------------------------------------------------------------------------
extern "C" void kernel_launch(void* const* d_in, const int* in_sizes, int n_in,
                              void* d_out, int out_size, void* d_ws, size_t ws_size,
                              hipStream_t stream) {
    const float* x      = (const float*)d_in[0];
    const int*   ctype  = (const int*)d_in[1];
    const int*   active = (const int*)d_in[2];
    const float* Wq = (const float*)d_in[3];   const float* bq = (const float*)d_in[4];
    const float* Wk = (const float*)d_in[5];   const float* bk = (const float*)d_in[6];
    const float* Wv = (const float*)d_in[7];   const float* bv = (const float*)d_in[8];
    const float* Wo = (const float*)d_in[9];   const float* bo = (const float*)d_in[10];
    const float* ct_bias = (const float*)d_in[11];
    const float* ct_key  = (const float*)d_in[12];
    const float* ln1g = (const float*)d_in[13]; const float* ln1b = (const float*)d_in[14];
    const float* ln2g = (const float*)d_in[15]; const float* ln2b = (const float*)d_in[16];
    const float* W1 = (const float*)d_in[17];  const float* b1 = (const float*)d_in[18];
    const float* W2 = (const float*)d_in[19];  const float* b2 = (const float*)d_in[20];
    float* out = (float*)d_out;

    // ---- workspace carve-out (256B aligned) ----
    char* ws = (char*)d_ws;
    size_t off = 0;
    auto take = [&](size_t bytes) -> char* {
        char* p = ws + off;
        off += (bytes + 255) & ~(size_t)255;
        return p;
    };
    _Float16* wq_t = (_Float16*)take((size_t)HID * HID * 2);       // transposed f16 weights
    _Float16* wk_t = (_Float16*)take((size_t)HID * HID * 2);
    _Float16* wv_t = (_Float16*)take((size_t)HID * HID * 2);
    _Float16* wo_t = (_Float16*)take((size_t)HID * HID * 2);
    _Float16* w1_t = (_Float16*)take((size_t)HID * 2 * HID * 2);
    _Float16* w2_t = (_Float16*)take((size_t)2 * HID * HID * 2);
    _Float16* normed = (_Float16*)take((size_t)MROWS * HID * 2);   // LN1, later LN2
    _Float16* qb     = (_Float16*)take((size_t)MROWS * HID * 2);
    _Float16* kb     = (_Float16*)take((size_t)MROWS * HID * 2);
    _Float16* vb     = (_Float16*)take((size_t)MROWS * HID * 2);
    _Float16* attnb  = (_Float16*)take((size_t)MROWS * HID * 2);
    _Float16* h1b    = qb;   // reuse q+k region (256MB contiguous) for FFN hidden

    // ---- 1) convert + transpose weights to f16 (Wt is N x K row-major) ----
    dim3 blk(256);
    auto convT = [&](const float* s, _Float16* d, int K, int N) {
        cca_convert_transpose_f16<<<dim3(K / 32, N / 32), blk, 0, stream>>>(s, d, K, N);
    };
    convT(Wq, wq_t, HID, HID);
    convT(Wk, wk_t, HID, HID);
    convT(Wv, wv_t, HID, HID);
    convT(Wo, wo_t, HID, HID);
    convT(W1, w1_t, HID, 2 * HID);
    convT(W2, w2_t, 2 * HID, HID);

    // ---- 2) LN1 ----
    cca_layernorm_f16<<<dim3(MROWS), blk, 0, stream>>>(x, ln1g, ln1b, normed);

    // ---- 3) Q / K / V projections (WMMA) ----
    dim3 g1k(HID / 128, MROWS / 128);       // N=1024
    dim3 g2k(2 * HID / 128, MROWS / 128);   // N=2048
    cca_gemm_wmma<0><<<g1k, blk, 0, stream>>>(normed, wq_t, bq, qb, nullptr, nullptr, nullptr, HID, HID);
    cca_gemm_wmma<1><<<g1k, blk, 0, stream>>>(normed, wk_t, bk, kb, nullptr, ct_key, ctype, HID, HID);
    cca_gemm_wmma<0><<<g1k, blk, 0, stream>>>(normed, wv_t, bv, vb, nullptr, nullptr, nullptr, HID, HID);

    // ---- 4) attention core ----
    cca_attention<<<dim3(BDIM * NH), blk, 0, stream>>>(qb, kb, vb, ct_bias, ctype, active, attnb);

    // ---- 5) output projection + residual -> d_out (fp32) ----
    cca_gemm_wmma<2><<<g1k, blk, 0, stream>>>(attnb, wo_t, bo, out, x, nullptr, nullptr, HID, HID);

    // ---- 6) LN2 ----
    cca_layernorm_f16<<<dim3(MROWS), blk, 0, stream>>>(out, ln2g, ln2b, normed);

    // ---- 7) FFN: h1 = elu(ln2 @ W1 + b1) ----
    cca_gemm_wmma<3><<<g2k, blk, 0, stream>>>(normed, w1_t, b1, h1b, nullptr, nullptr, nullptr, 2 * HID, HID);

    // ---- 8) d_out += h1 @ W2 + b2 ----
    cca_gemm_wmma<2><<<g1k, blk, 0, stream>>>(h1b, w2_t, b2, out, out, nullptr, nullptr, HID, 2 * HID);

    (void)in_sizes; (void)n_in; (void)out_size; (void)ws_size;
}
